// DynamicDeformableConv2d_26663156974170
// MI455X (gfx1250) — matrix operand order repaired and checked by reading, not compiled, not run on hardware
//
#include <hip/hip_runtime.h>

// ---------------------------------------------------------------------------
// Deformable conv2d for MI455X (gfx1250, wave32, WMMA).
//   B=4, C=64, H=W=128, O=64, K=3  (PAD=1, STRIDE=1)
//
// Fully fused design:
//   k_wcvt   : repack weight (64x576) and offset_w (18x576, zero-padded to
//              32x576) fp32 -> bf16 row-major A-matrices in workspace.
//   k_deform : per 16-pixel strip --
//     phase 0: im2col of raw x (3x3 taps) into LDS B-matrix [576][16] bf16
//     phase 1: 18x16 offset field via WMMA (waves 0-1, A = offset_w)
//     phase 2: bilinear positions/weights from LDS offsets
//     phase 3: deformed bilinear samples overwrite the LDS B-matrix
//     phase 4: main GEMM D[64 ch][16 pix] via WMMA (all 4 waves), K=576
//   No global intermediate for offsets: saves ~9.4 MB HBM round-trip and
//   replaces a ~1.4 GFLOP scalar VALU conv with matrix-pipe work.
// ---------------------------------------------------------------------------

typedef __attribute__((ext_vector_type(16))) __bf16 v16bf;
typedef __attribute__((ext_vector_type(8)))  __bf16 v8bf;
typedef __attribute__((ext_vector_type(8)))  float  v8f;

#define B_   4
#define C_   64
#define H_   128
#define W_   128
#define O_   64
#define K2_  9
#define CK_  576          // C_ * K2_
#define HP_  130          // H + 2*PAD
#define NOFF 18           // 2*K*K offset channels
#define KSTEPS (CK_ / 32) // 18 WMMA K-steps
#define ROWS 584          // LDS row stride in bf16 elems (576 + 8 pad):
                          // 584*2B = 292 dwords/row; 292 mod 64 = 36,
                          // gcd(36,64)=4 -> 16 distinct row-start banks,
                          // 16 lanes x 4 dwords = all 64 banks, conflict-free.

__device__ __forceinline__ unsigned short f2bf(float f) {
    unsigned int u = __builtin_bit_cast(unsigned int, f);
    u += 0x7FFFu + ((u >> 16) & 1u);          // round-to-nearest-even
    return (unsigned short)(u >> 16);
}

// ---- repack both weight arrays to bf16 WMMA A-matrices ---------------------
//   wbf : [64][576]  <- weight  (O,C,3,3) flat is already [o][c*9+k]
//   owbf: [32][576]  <- offset_w rows 0..17, rows 18..31 zero
__global__ void k_wcvt(const float* __restrict__ w,
                       const float* __restrict__ ow,
                       unsigned short* __restrict__ wbf,
                       unsigned short* __restrict__ owbf) {
    int i = blockIdx.x * blockDim.x + threadIdx.x;
    if (i < O_ * CK_) wbf[i] = f2bf(w[i]);
    if (i < 32 * CK_) owbf[i] = (i < NOFF * CK_) ? f2bf(ow[i]) : (unsigned short)0;
}

// ---- fused offset-conv + bilinear sampling + WMMA GEMM ---------------------
// grid: B * H * (W/16) blocks, 128 threads (4 wave32).
__global__ void __launch_bounds__(128)
k_deform(const float* __restrict__ x,
         const unsigned short* __restrict__ wbf,
         const unsigned short* __restrict__ owbf,
         const float* __restrict__ bias,
         const float* __restrict__ ob,
         float* __restrict__ out) {
    __shared__ unsigned short s_val[16 * ROWS];  // B-matrix: [pixel][c*9+k] bf16
    __shared__ float s_off[NOFF * 16];           // offset field [chan][pixel]
    __shared__ int   s_h0[144], s_w0[144];       // 16 pixels x 9 taps
    __shared__ float s_wq[144][4];               // wh0, wh1, ww0, ww1

    const int tid = threadIdx.x;
    const int bid = blockIdx.x;
    const int wt = bid & 7;                // 8 tiles of 16 pixels per row
    const int ho = (bid >> 3) & (H_ - 1);
    const int b  = bid >> 10;
    const int wo_base = wt * 16;

    const int wave = tid >> 5;
    const int lane = tid & 31;
    const int n    = lane & 15;            // pixel column / A-row within tile
    const int sel  = (lane >> 4) & 1;      // half-wave select

    const float* xb = x + (size_t)b * C_ * H_ * W_;

    // --- phase 0: im2col of raw x into LDS B-matrix (zero-padded border) ----
    for (int e = tid; e < C_ * 144; e += 128) {
        int c   = e / 144;
        int idx = e - c * 144;
        int p = idx / 9, k = idx - p * 9;
        int kh = k / 3, kw = k - kh * 3;
        int ih = ho + kh - 1;
        int iw = wo_base + p + kw - 1;
        float v = ((unsigned)ih < (unsigned)H_ && (unsigned)iw < (unsigned)W_)
                      ? xb[c * (H_ * W_) + ih * W_ + iw] : 0.0f;
        s_val[p * ROWS + c * 9 + k] = f2bf(v);
    }
    __syncthreads();

    // --- phase 1: offset field = offset_w GEMM (waves 0-1, WMMA) ------------
    if (wave < 2) {
        v8f oacc = {};
        const unsigned short* orow = owbf + (size_t)(wave * 16 + n) * CK_;
        for (int kb = 0; kb < KSTEPS; ++kb) {
            const int k0 = kb * 32;
            v8bf a_lo = *(const v8bf*)(orow + k0 + sel * 8);
            v8bf a_hi = *(const v8bf*)(orow + k0 + 16 + sel * 8);
            v16bf A = __builtin_shufflevector(a_lo, a_hi,
                        0,1,2,3,4,5,6,7,8,9,10,11,12,13,14,15);
            const unsigned short* vrow = s_val + n * ROWS + k0 + sel * 8;
            v8bf b_lo = *(const v8bf*)(vrow);
            v8bf b_hi = *(const v8bf*)(vrow + 16);
            v16bf Bm = __builtin_shufflevector(b_lo, b_hi,
                        0,1,2,3,4,5,6,7,8,9,10,11,12,13,14,15);
            oacc = __builtin_amdgcn_wmma_f32_16x16x32_bf16(
                       false, A, false, Bm, (short)0, oacc, false, false);
        }
        // D: lane n = pixel; VGPR j = channel (wave*16 + sel*8 + j)
#pragma unroll
        for (int j = 0; j < 8; ++j) {
            int ch = wave * 16 + sel * 8 + j;
            if (ch < NOFF) s_off[ch * 16 + n] = oacc[j] + ob[ch];
        }
    }
    __syncthreads();

    // --- phase 2: bilinear positions/weights (shared across all channels) ---
    for (int idx = tid; idx < 144; idx += 128) {
        int p = idx / 9, k = idx - p * 9;
        int kh = k / 3, kw = k - kh * 3;
        int wo = wo_base + p;
        float off_h = s_off[k * 16 + p];
        float off_w = s_off[(9 + k) * 16 + p];
        float ph = (float)kh + off_h + (float)ho + 1.0f;   // +PAD
        float pw = (float)kw + off_w + (float)wo + 1.0f;
        ph = fminf(fmaxf(ph, 0.0f), (float)(HP_ - 1));
        pw = fminf(fmaxf(pw, 0.0f), (float)(HP_ - 1));
        int h0 = (int)floorf(ph);
        int w0 = (int)floorf(pw);
        int h1 = min(h0 + 1, HP_ - 1);
        int w1 = min(w0 + 1, HP_ - 1);
        s_h0[idx] = h0;  s_w0[idx] = w0;
        s_wq[idx][0] = (float)h1 - ph;   // wh0
        s_wq[idx][1] = ph - (float)h0;   // wh1
        s_wq[idx][2] = (float)w1 - pw;   // ww0
        s_wq[idx][3] = pw - (float)w0;   // ww1
    }
    __syncthreads();

    // --- phase 3: deformed bilinear samples overwrite the LDS B-matrix ------
    for (int e = tid; e < C_ * 144; e += 128) {
        int c   = e / 144;
        int idx = e - c * 144;
        int p = idx / 9, k = idx - p * 9;
        const float* xc = xb + c * (H_ * W_);
        int h0 = s_h0[idx], w0 = s_w0[idx];
        int h1 = min(h0 + 1, HP_ - 1), w1 = min(w0 + 1, HP_ - 1);
        float wh0 = s_wq[idx][0], wh1 = s_wq[idx][1];
        float ww0 = s_wq[idx][2], ww1 = s_wq[idx][3];
        auto samp = [&](int h, int w) -> float {   // zero-padded border
            return (h >= 1 && h <= H_ && w >= 1 && w <= W_)
                       ? xc[(h - 1) * W_ + (w - 1)] : 0.0f;
        };
        float v = samp(h0, w0) * (wh0 * ww0) + samp(h0, w1) * (wh0 * ww1)
                + samp(h1, w0) * (wh1 * ww0) + samp(h1, w1) * (wh1 * ww1);
        s_val[p * ROWS + c * 9 + k] = f2bf(v);
    }
    __syncthreads();

    // --- phase 4: main GEMM over K = 576 (all 4 waves) ----------------------
    v8f acc = {};
    const unsigned short* wrow = wbf + (size_t)(wave * 16 + n) * CK_;
    for (int kb = 0; kb < KSTEPS; ++kb) {
        const int k0 = kb * 32;
        // A fragment (16-bit A 16x32 layout): lanes 0-15 K{0..7,16..23},
        // lanes 16-31 K{8..15,24..31}; two 16B loads, 16B-aligned.
        v8bf a_lo = *(const v8bf*)(wrow + k0 + sel * 8);
        v8bf a_hi = *(const v8bf*)(wrow + k0 + 16 + sel * 8);
        v16bf A = __builtin_shufflevector(a_lo, a_hi,
                    0,1,2,3,4,5,6,7,8,9,10,11,12,13,14,15);
        // B fragment (32x16): lane n = column, K{0..7,16..23} lanes 0-15,
        // K{8..15,24..31} lanes 16-31; two ds_load_b128.
        const unsigned short* vrow = s_val + n * ROWS + k0 + sel * 8;
        v8bf b_lo = *(const v8bf*)(vrow);
        v8bf b_hi = *(const v8bf*)(vrow + 16);
        v16bf Bm = __builtin_shufflevector(b_lo, b_hi,
                    0,1,2,3,4,5,6,7,8,9,10,11,12,13,14,15);
        acc = __builtin_amdgcn_wmma_f32_16x16x32_bf16(
                  false, A, false, Bm, (short)0, acc, false, false);
    }

    // D layout: lane 0-15 -> N=lane, VGPR j -> M=j; lanes 16-31 -> M=8+j.
    const int wo = wo_base + n;
#pragma unroll
    for (int j = 0; j < 8; ++j) {
        int ch = wave * 16 + sel * 8 + j;
        out[(((size_t)b * O_ + ch) * H_ + ho) * W_ + wo] = acc[j] + bias[ch];
    }
}

// ---------------------------------------------------------------------------
extern "C" void kernel_launch(void* const* d_in, const int* in_sizes, int n_in,
                              void* d_out, int out_size, void* d_ws, size_t ws_size,
                              hipStream_t stream) {
    const float* x        = (const float*)d_in[0];  // (4,64,128,128)
    const float* weight   = (const float*)d_in[1];  // (64,64,3,3)
    const float* bias     = (const float*)d_in[2];  // (64,)
    const float* offset_w = (const float*)d_in[3];  // (18,64,3,3)
    const float* offset_b = (const float*)d_in[4];  // (18,)
    float* out = (float*)d_out;                     // (4,64,128,128)

    // workspace: [wbf bf16 64x576 | owbf bf16 32x576]  (~110 KB total)
    unsigned short* wbf  = (unsigned short*)d_ws;
    unsigned short* owbf = wbf + (size_t)O_ * CK_;

    k_wcvt<<<(O_ * CK_ + 255) / 256, 256, 0, stream>>>(
        weight, offset_w, wbf, owbf);
    k_deform<<<B_ * H_ * (W_ / 16), 128, 0, stream>>>(
        x, wbf, owbf, bias, offset_b, out);
}
